// MoEEP_20624432955729
// MI455X (gfx1250) — compile-verified, matrix operand order checked by reading, not run on hardware
//
#include <hip/hip_runtime.h>
#include <hip/hip_bf16.h>

// ---------------------------------------------------------------------------
// MoE forward for MI455X (gfx1250): bf16 WMMA GEMMs, fp32 accumulate.
// N=4096 tokens, D=2048, H=8192, E=8 experts, top-2, capacity 2560.
// 256x128 tiles (512 thr / 16 waves), async global->LDS for bf16 A tiles.
// ---------------------------------------------------------------------------

#define NTOK 4096
#define DDIM 2048
#define HDIM 8192
#define NEXP 8
#define CAP  2560   // int(1.25 * 4096 / 2)

typedef __bf16 bf16_t;
typedef bf16_t bf16x16 __attribute__((ext_vector_type(16)));
typedef float  f32x8   __attribute__((ext_vector_type(8)));
typedef unsigned int u32x4 __attribute__((ext_vector_type(4)));
typedef int v2i __attribute__((ext_vector_type(2)));

union Frag {
    bf16x16 v;
    u32x4   q[2];
};

#define AS1 __attribute__((address_space(1)))
#define AS3 __attribute__((address_space(3)))

#if defined(__has_builtin)
#if __has_builtin(__builtin_amdgcn_global_load_async_to_lds_b64)
#define HAVE_ASYNC_LDS 1
#endif
#endif

__device__ __forceinline__ void copy8_to_lds(void* lds_dst, const void* gsrc) {
#if defined(HAVE_ASYNC_LDS)
    __builtin_amdgcn_global_load_async_to_lds_b64(
        (AS1 v2i*)(void*)gsrc, (AS3 v2i*)lds_dst, 0, 0);
#else
    *(unsigned long long*)lds_dst = *(const unsigned long long*)gsrc;
#endif
}

__device__ __forceinline__ void async_lds_wait() {
#if defined(HAVE_ASYNC_LDS)
#if __has_builtin(__builtin_amdgcn_s_wait_asynccnt)
    __builtin_amdgcn_s_wait_asynccnt(0);
#else
    asm volatile("s_wait_asynccnt 0" ::: "memory");
#endif
#endif
}

// fp32 -> bf16 round-to-nearest-even (bit pattern as ushort)
__device__ __forceinline__ unsigned short f2bf(float f) {
    unsigned int u = __float_as_uint(f);
    u += 0x7FFFu + ((u >> 16) & 1u);
    return (unsigned short)(u >> 16);
}

// branch-free tanh-form GELU (error << bf16 matmul noise)
__device__ __forceinline__ float gelu_f(float v) {
    float u  = 0.7978845608028654f * (v + 0.044715f * v * v * v);
    float e  = __expf(2.0f * u);
    float th = 1.0f - 2.0f / (e + 1.0f);     // tanh(u), saturates correctly
    return 0.5f * v * (1.0f + th);
}

// ---------------------------------------------------------------------------
// Pre-convert x to bf16 once per call (halves gemm1 A-tile traffic).
// ---------------------------------------------------------------------------
__global__ void cvt_kernel(const float* __restrict__ src,
                           unsigned short* __restrict__ dst, int n) {
    int i = blockIdx.x * blockDim.x + threadIdx.x;
    if (i < n) dst[i] = f2bf(src[i]);
}

// ---------------------------------------------------------------------------
// Router: logits = x @ rw + rb, softmax, top-2 (lower index wins ties).
// ---------------------------------------------------------------------------
__global__ __launch_bounds__(256) void router_kernel(
    const float* __restrict__ x, const float* __restrict__ rw,
    const float* __restrict__ rb, float* __restrict__ probs,
    int* __restrict__ topi, float* __restrict__ topw) {
    __shared__ float sRW[1024 * NEXP];   // 32 KB staging of router weights
    const int t   = threadIdx.x;
    const int tok = blockIdx.x * 256 + t;

    float acc[NEXP];
#pragma unroll
    for (int e = 0; e < NEXP; ++e) acc[e] = rb[e];

    for (int d0 = 0; d0 < DDIM; d0 += 1024) {
        for (int i = t; i < 1024 * NEXP; i += 256) sRW[i] = rw[d0 * NEXP + i];
        __syncthreads();
        for (int dd = 0; dd < 1024; ++dd) {
            float xv = x[(size_t)tok * DDIM + d0 + dd];
#pragma unroll
            for (int e = 0; e < NEXP; ++e)
                acc[e] = fmaf(xv, sRW[dd * NEXP + e], acc[e]);
        }
        __syncthreads();
    }

    float mx = acc[0];
#pragma unroll
    for (int e = 1; e < NEXP; ++e) mx = fmaxf(mx, acc[e]);
    float p[NEXP], s = 0.f;
#pragma unroll
    for (int e = 0; e < NEXP; ++e) { p[e] = expf(acc[e] - mx); s += p[e]; }
    float inv = 1.f / s;
#pragma unroll
    for (int e = 0; e < NEXP; ++e) { p[e] *= inv; probs[tok * NEXP + e] = p[e]; }

    int i0 = 0; float v0 = p[0];
#pragma unroll
    for (int e = 1; e < NEXP; ++e) if (p[e] > v0) { v0 = p[e]; i0 = e; }
    int i1 = -1; float v1 = -1.f;
#pragma unroll
    for (int e = 0; e < NEXP; ++e) if (e != i0 && p[e] > v1) { v1 = p[e]; i1 = e; }

    topi[tok * 2 + 0] = i0; topi[tok * 2 + 1] = i1;
    topw[tok * 2 + 0] = v0; topw[tok * 2 + 1] = v1;
}

// ---------------------------------------------------------------------------
// Capacity assignment: one wave32 per expert, in-order ballot prefix scan.
// ---------------------------------------------------------------------------
__global__ __launch_bounds__(32) void assign_kernel(
    const int* __restrict__ topi, const float* __restrict__ topw,
    int* __restrict__ tokIdx, float* __restrict__ tokW, int* __restrict__ counts) {
    const int e    = blockIdx.x;
    const int lane = threadIdx.x;

    for (int sI = lane; sI < CAP; sI += 32) {
        tokIdx[e * CAP + sI] = 0;
        tokW[e * CAP + sI]   = 0.f;
    }
    // single wave: per-wave stores are ordered with stores

    int base = 0;
    for (int t0 = 0; t0 < NTOK; t0 += 32) {
        int tok = t0 + lane;
        int a = topi[tok * 2 + 0], b = topi[tok * 2 + 1];
        float w = (a == e) ? topw[tok * 2 + 0] : ((b == e) ? topw[tok * 2 + 1] : 0.f);
        bool sel = (a == e) || (b == e);
        unsigned mask = (unsigned)__ballot(sel);
        int slot = base + __popc(mask & ((1u << lane) - 1u));
        if (sel && slot < CAP) {
            tokIdx[e * CAP + slot] = tok;
            tokW[e * CAP + slot]   = w;
        }
        base += __popc(mask);
    }
    if (lane == 0) counts[e] = base;
}

// ---------------------------------------------------------------------------
// GEMM1: h1 = GELU(gather(xbf) @ w1 + b1)  [CAP x HDIM], bf16 output.
// 256x128 tile / 512-thread block, 16 waves (8M x 2N), wave = 32x64.
// ---------------------------------------------------------------------------
__global__ __launch_bounds__(512) void gemm1_kernel(
    const unsigned short* __restrict__ xbf, const float* __restrict__ w1,
    const float* __restrict__ b1, const int* __restrict__ toks,
    unsigned short* __restrict__ h1) {
    __shared__ __align__(16) unsigned short lA[256][40];  // 80 B row stride
    __shared__ __align__(16) unsigned short lB[128][40];
    __shared__ int sTok[256];

    const int t  = threadIdx.x;
    const int n0 = blockIdx.x * 128;
    const int m0 = blockIdx.y * 256;
    if (t < 256) sTok[t] = toks[m0 + t];
    __syncthreads();

    const int lane = t & 31;
    const int wv   = t >> 5;              // 0..15
    const int wm   = (wv >> 1) * 32;      // 0..224
    const int wn   = (wv & 1) * 64;
    const int lr   = lane & 15;
    const int lg   = lane >> 4;

    f32x8 acc[2][4];
#pragma unroll
    for (int i = 0; i < 2; ++i)
#pragma unroll
        for (int j = 0; j < 4; ++j) acc[i][j] = (f32x8){};

    const int bn  = t & 127;
    const int bk0 = (t >> 7) * 8;         // 0,8,16,24

    for (int k0 = 0; k0 < DDIM; k0 += 32) {
        // A tile: 256 rows x 32 k of bf16, 8 B (4-elem) copies at 8-elem offsets, async to LDS
#pragma unroll
        for (int i = 0; i < 2; ++i) {
            int id  = t + 512 * i;        // 0..1023
            int row = id >> 2, ch = id & 3;
            const unsigned short* src =
                xbf + (size_t)sTok[row] * DDIM + k0 + ch * 8;
            copy8_to_lds(&lA[row][ch * 8], src);
        }
        // B tile: w1 [D,H] row-major; stage transposed [n][k], fp32 -> bf16
#pragma unroll
        for (int kk = 0; kk < 8; ++kk) {
            int k = bk0 + kk;
            lB[bn][k] = f2bf(w1[(size_t)(k0 + k) * HDIM + n0 + bn]);
        }
        async_lds_wait();
        __syncthreads();

        Frag a[2], bfr[4];
#pragma unroll
        for (int mi = 0; mi < 2; ++mi) {
            int row = wm + mi * 16 + lr;
            a[mi].q[0] = *(const u32x4*)&lA[row][lg * 8];
            a[mi].q[1] = *(const u32x4*)&lA[row][16 + lg * 8];
        }
#pragma unroll
        for (int ni = 0; ni < 4; ++ni) {
            int col = wn + ni * 16 + lr;
            bfr[ni].q[0] = *(const u32x4*)&lB[col][lg * 8];
            bfr[ni].q[1] = *(const u32x4*)&lB[col][16 + lg * 8];
        }
#pragma unroll
        for (int mi = 0; mi < 2; ++mi)
#pragma unroll
            for (int ni = 0; ni < 4; ++ni)
                acc[mi][ni] = __builtin_amdgcn_wmma_f32_16x16x32_bf16(
                    false, a[mi].v, false, bfr[ni].v, (short)0, acc[mi][ni],
                    false, false);
        __syncthreads();
    }

#pragma unroll
    for (int mi = 0; mi < 2; ++mi) {
#pragma unroll
        for (int ni = 0; ni < 4; ++ni) {
            int gn = n0 + wn + ni * 16 + lr;
            float bias = b1[gn];
#pragma unroll
            for (int r = 0; r < 8; ++r) {
                int gm = m0 + wm + mi * 16 + lg * 8 + r;
                float v = acc[mi][ni][r] + bias;
                h1[(size_t)gm * HDIM + gn] = f2bf(gelu_f(v));
            }
        }
    }
}

// ---------------------------------------------------------------------------
// GEMM2: y = h1 @ w2 + b2; scatter final[tok] += w * y (atomic f32 adds;
// <= 2 commutative contributions per element -> deterministic).
// ---------------------------------------------------------------------------
__global__ __launch_bounds__(512) void gemm2_kernel(
    const unsigned short* __restrict__ h1, const float* __restrict__ w2,
    const float* __restrict__ b2, const int* __restrict__ toks,
    const float* __restrict__ tw, float* __restrict__ out) {
    __shared__ __align__(16) unsigned short lA[256][40];
    __shared__ __align__(16) unsigned short lB[128][40];
    __shared__ int   sTok[256];
    __shared__ float sW[256];

    const int t  = threadIdx.x;
    const int n0 = blockIdx.x * 128;
    const int m0 = blockIdx.y * 256;
    if (t < 256) { sTok[t] = toks[m0 + t]; sW[t] = tw[m0 + t]; }
    __syncthreads();

    const int lane = t & 31;
    const int wv   = t >> 5;
    const int wm   = (wv >> 1) * 32;
    const int wn   = (wv & 1) * 64;
    const int lr   = lane & 15;
    const int lg   = lane >> 4;

    f32x8 acc[2][4];
#pragma unroll
    for (int i = 0; i < 2; ++i)
#pragma unroll
        for (int j = 0; j < 4; ++j) acc[i][j] = (f32x8){};

    const int bn  = t & 127;
    const int bk0 = (t >> 7) * 8;

    for (int k0 = 0; k0 < HDIM; k0 += 32) {
        // A tile: h1 bf16 straight copy, async to LDS
#pragma unroll
        for (int i = 0; i < 2; ++i) {
            int id  = t + 512 * i;
            int row = id >> 2, ch = id & 3;
            const unsigned short* src =
                h1 + (size_t)(m0 + row) * HDIM + k0 + ch * 8;
            copy8_to_lds(&lA[row][ch * 8], src);
        }
        // B tile: w2 [H,D] row-major; stage transposed [n][k], fp32 -> bf16
#pragma unroll
        for (int kk = 0; kk < 8; ++kk) {
            int k = bk0 + kk;
            lB[bn][k] = f2bf(w2[(size_t)(k0 + k) * DDIM + n0 + bn]);
        }
        async_lds_wait();
        __syncthreads();

        Frag a[2], bfr[4];
#pragma unroll
        for (int mi = 0; mi < 2; ++mi) {
            int row = wm + mi * 16 + lr;
            a[mi].q[0] = *(const u32x4*)&lA[row][lg * 8];
            a[mi].q[1] = *(const u32x4*)&lA[row][16 + lg * 8];
        }
#pragma unroll
        for (int ni = 0; ni < 4; ++ni) {
            int col = wn + ni * 16 + lr;
            bfr[ni].q[0] = *(const u32x4*)&lB[col][lg * 8];
            bfr[ni].q[1] = *(const u32x4*)&lB[col][16 + lg * 8];
        }
#pragma unroll
        for (int mi = 0; mi < 2; ++mi)
#pragma unroll
            for (int ni = 0; ni < 4; ++ni)
                acc[mi][ni] = __builtin_amdgcn_wmma_f32_16x16x32_bf16(
                    false, a[mi].v, false, bfr[ni].v, (short)0, acc[mi][ni],
                    false, false);
        __syncthreads();
    }

#pragma unroll
    for (int mi = 0; mi < 2; ++mi) {
#pragma unroll
        for (int ni = 0; ni < 4; ++ni) {
            int gn = n0 + wn + ni * 16 + lr;
            float bias = b2[gn];
#pragma unroll
            for (int r = 0; r < 8; ++r) {
                int lm  = wm + mi * 16 + lg * 8 + r;
                float w = sW[lm];
                if (w != 0.f) {
                    int tok = sTok[lm];
                    float y = acc[mi][ni][r] + bias;
                    atomicAdd(&out[(size_t)tok * DDIM + gn], w * y);
                }
            }
        }
    }
}

// ---------------------------------------------------------------------------
// Aux loss: balance + importance, deterministic fixed-tree reduction.
// ---------------------------------------------------------------------------
__global__ __launch_bounds__(256) void aux_kernel(
    const float* __restrict__ probs, const int* __restrict__ counts,
    float* __restrict__ outScalar) {
    __shared__ float simp[NEXP];
    const int t = threadIdx.x;
    const int e = t >> 5, lane = t & 31;
    float s = 0.f;
    for (int n = lane; n < NTOK; n += 32) s += probs[n * NEXP + e];
#pragma unroll
    for (int off = 16; off > 0; off >>= 1) s += __shfl_down(s, off, 32);
    if (lane == 0) simp[e] = s;
    __syncthreads();
    if (t == 0) {
        float bal = 0.f, imp2 = 0.f;
        for (int i = 0; i < NEXP; ++i) {
            float density = simp[i] / (float)NTOK;
            float usage   = (float)counts[i] / (float)NTOK;
            bal  += density * usage;
            imp2 += simp[i] * simp[i];
        }
        *outScalar = bal * (float)NEXP + imp2 / (float)NEXP;
    }
}

__global__ void zero_kernel(float* __restrict__ p, int n) {
    int i = blockIdx.x * blockDim.x + threadIdx.x;
    if (i < n) p[i] = 0.f;
}

// ---------------------------------------------------------------------------
extern "C" void kernel_launch(void* const* d_in, const int* in_sizes, int n_in,
                              void* d_out, int out_size, void* d_ws, size_t ws_size,
                              hipStream_t stream) {
    const float* x  = (const float*)d_in[0];
    const float* rw = (const float*)d_in[1];
    const float* rb = (const float*)d_in[2];
    const float* w1 = (const float*)d_in[3];
    const float* b1 = (const float*)d_in[4];
    const float* w2 = (const float*)d_in[5];
    const float* b2 = (const float*)d_in[6];
    float* out = (float*)d_out;

    // workspace layout
    char* ws = (char*)d_ws;
    float* probs  = (float*)(ws + 0);                 // N*E       = 128 KB
    int*   topi   = (int*)  (ws + 131072);            // N*2       =  32 KB
    float* topw   = (float*)(ws + 163840);            // N*2       =  32 KB
    int*   tokIdx = (int*)  (ws + 196608);            // E*CAP     =  80 KB
    float* tokW   = (float*)(ws + 278528);            // E*CAP     =  80 KB
    int*   counts = (int*)  (ws + 360448);            // E
    unsigned short* xbf = (unsigned short*)(ws + (1ull  << 20)); // N*D bf16 = 16 MB
    unsigned short* h1  = (unsigned short*)(ws + (20ull << 20)); // CAP*H bf16 = 40 MB

    const int outN = NTOK * DDIM + 1;
    zero_kernel<<<(outN + 255) / 256, 256, 0, stream>>>(out, outN);
    cvt_kernel<<<(NTOK * DDIM + 255) / 256, 256, 0, stream>>>(x, xbf, NTOK * DDIM);
    router_kernel<<<NTOK / 256, 256, 0, stream>>>(x, rw, rb, probs, topi, topw);
    assign_kernel<<<NEXP, 32, 0, stream>>>(topi, topw, tokIdx, tokW, counts);

    for (int e = 0; e < NEXP; ++e) {
        gemm1_kernel<<<dim3(HDIM / 128, CAP / 256), 512, 0, stream>>>(
            xbf, w1 + (size_t)e * DDIM * HDIM, b1 + (size_t)e * HDIM,
            tokIdx + (size_t)e * CAP, h1);
        gemm2_kernel<<<dim3(DDIM / 128, CAP / 256), 512, 0, stream>>>(
            h1, w2 + (size_t)e * HDIM * DDIM, b2 + (size_t)e * DDIM,
            tokIdx + (size_t)e * CAP, tokW + (size_t)e * CAP, out);
    }

    aux_kernel<<<1, 256, 0, stream>>>(probs, counts, out + (size_t)NTOK * DDIM);
}
